// Conv2d_61117384622815
// MI455X (gfx1250) — matrix operand order repaired and checked by reading, not compiled, not run on hardware
//
#include <hip/hip_runtime.h>
#include <hip/hip_bf16.h>

typedef __attribute__((ext_vector_type(16))) __bf16 v16bf;
typedef __attribute__((ext_vector_type(8)))  __bf16 v8bf;
typedef __attribute__((ext_vector_type(8)))  float  v8f;
typedef __attribute__((ext_vector_type(4)))  unsigned int v4u;
typedef __attribute__((ext_vector_type(8)))  int    v8i;
typedef __attribute__((ext_vector_type(4)))  int    v4i;

#define C_IN   32
#define C_OUT  32
#define HW     224
#define NTAPS  9
#define LDS_W  226                       // 224 + 2 halo (pad=1 each side)
#define W_ELEMS (C_OUT * NTAPS * C_IN)   // 9216 bf16 = 18432 B
#define X_ELEMS (3 * LDS_W * C_IN)       // 21696 bf16 = 43392 B

// ---------------------------------------------------------------------------
// Kernel 1: repack fp32 weights [oc][c][r][s] -> bf16 [oc][tap][c] in d_ws.
// ---------------------------------------------------------------------------
__global__ void repack_weights(const float* __restrict__ W,
                               __bf16* __restrict__ Wp) {
    int i = blockIdx.x * blockDim.x + threadIdx.x;
    if (i < W_ELEMS) {
        int c   = i & 31;
        int tap = (i >> 5) % NTAPS;
        int oc  = i / (NTAPS * C_IN);
        int r = tap / 3, s = tap % 3;
        Wp[i] = (__bf16)W[((oc * C_IN + c) * 3 + r) * 3 + s];
    }
}

// ---------------------------------------------------------------------------
// Kernel 2: implicit-GEMM conv via v_wmma_f32_16x16x32_bf16.
// Block = one (n, h) output row. 224 threads = 7 waves.
// Weights staged LDS-side by the Tensor Data Mover (1-D 18KB tile);
// input rows staged with float4 global loads + bf16 conversion.
// ---------------------------------------------------------------------------
__global__ __launch_bounds__(224)
void conv_wmma_bf16(const float* __restrict__ X,
                    const __bf16* __restrict__ Wp,
                    const float* __restrict__ bias,
                    float* __restrict__ Y) {
    __shared__ __attribute__((aligned(16))) __bf16 sW[W_ELEMS]; // [oc][tap][c]
    __shared__ __attribute__((aligned(16))) __bf16 sX[X_ELEMS]; // [row][w][c]

    const int tid = threadIdx.x;
    const int n = blockIdx.x / HW;
    const int h = blockIdx.x % HW;

    // --- stage weights via TDM: contiguous 18432B global -> LDS ------------
    if (tid < 32) {                       // wave 0 only (uniform branch)
        const unsigned long long ga = (unsigned long long)(uintptr_t)Wp;
        const unsigned int lds_off   = (unsigned int)(uintptr_t)sW;

        // D# group0: count=1 | lds_addr | global_addr[56:0] | type=2
        v4u g0;
        g0[0] = 1u;                                   // count=1, no gather
        g0[1] = lds_off;                              // LDS byte address
        g0[2] = (unsigned int)(ga & 0xFFFFFFFFu);     // global_addr[31:0]
        g0[3] = (unsigned int)((ga >> 32) & 0x1FFFFFFu) | 0x80000000u; // type=2

        // D# group1: data_size=8B (code 3); 1-D tensor/tile of 2304 elems
        const unsigned int n8 = (W_ELEMS * 2) / 8;    // 2304 8-byte elems
        v8i g1;
        g1[0] = (int)(3u << 16);                      // wg_mask=0, data_size=3
        g1[1] = (int)((n8 & 0xFFFFu) << 16);          // tensor_dim0[15:0]
        g1[2] = (int)(1u << 16);                      // tensor_dim1 = 1
        g1[3] = (int)((n8 & 0xFFFFu) << 16);          // tile_dim0 = 2304
        g1[4] = 0;                                    // tile_dim1/2 = 0
        g1[5] = (int)n8;                              // tensor_dim0_stride
        g1[6] = 0;
        g1[7] = 0;

        v4i g2 = {0, 0, 0, 0};
        v4i g3 = {0, 0, 0, 0};
#if defined(__clang_major__) && (__clang_major__ >= 23)
        v8i g4 = {0, 0, 0, 0, 0, 0, 0, 0};
        __builtin_amdgcn_tensor_load_to_lds(g0, g1, g2, g3, g4, 0);
#else
        __builtin_amdgcn_tensor_load_to_lds(g0, g1, g2, g3, 0);
#endif
        __builtin_amdgcn_s_wait_tensorcnt(0);
    }

    // --- zero the halo columns (w=0 and w=225) for all 96 (c,row) rows -----
    for (int i = tid; i < 96 * 2; i += 224) {
        int rowc = i >> 1;                // 0..95 = c*3+row
        int c   = rowc / 3;
        int row = rowc % 3;
        int w   = (i & 1) ? (LDS_W - 1) : 0;
        sX[(row * LDS_W + w) * C_IN + c] = (__bf16)0.0f;
    }

    // --- stage 3 input rows, float4 loads, fp32 -> bf16, c-innermost -------
    // 96 (c,row) rows x 56 float4 segments = 5376 vector loads.
    for (int j = tid; j < 96 * (HW / 4); j += 224) {
        int q    = j % (HW / 4);          // float4 index within row
        int rowc = j / (HW / 4);
        int c    = rowc / 3;
        int row  = rowc % 3;
        int hin  = h + row - 1;

        float4 v = make_float4(0.f, 0.f, 0.f, 0.f);
        if ((unsigned)hin < (unsigned)HW) {
            const float4* src = (const float4*)(X + ((n * C_IN + c) * HW + hin) * HW);
            v = src[q];
        }
        __bf16* dst = &sX[(row * LDS_W + 1 + q * 4) * C_IN + c];
        dst[0 * C_IN] = (__bf16)v.x;
        dst[1 * C_IN] = (__bf16)v.y;
        dst[2 * C_IN] = (__bf16)v.z;
        dst[3 * C_IN] = (__bf16)v.w;
    }
    __syncthreads();

    const int lane = tid & 31;
    const int wave = tid >> 5;       // 0..6
    const int half = lane >> 4;      // 0: lanes 0-15, 1: lanes 16-31
    const int l    = lane & 15;

    const int wb0 = wave * 32;       // first 16-pixel tile base
    const int wb1 = wave * 32 + 16;  // second tile base

    // bf16 A (16x32) K order chosen here (B matches): lane<16 -> K=0..15, lane>=16 ->
    // K=16..31; K == input channel for this tap.
    const int ca0 = half ? 16 : 0;
    const int ca1 = half ? 24 : 8;
    // bf16 B (32x16), same K order as A: lane<16 -> K=0..15, lane>=16 -> K=16..31, N = lane%16.
    const int cb = half * 16;

    v8f acc00 = {}; // oc 0..15  x tile0
    v8f acc10 = {}; // oc 16..31 x tile0
    v8f acc01 = {}; // oc 0..15  x tile1
    v8f acc11 = {}; // oc 16..31 x tile1

    #pragma unroll
    for (int tap = 0; tap < NTAPS; ++tap) {
        const int r = tap / 3, s = tap % 3;

        const __bf16* wp0 = &sW[(l        * NTAPS + tap) * C_IN];
        const __bf16* wp1 = &sW[((l + 16) * NTAPS + tap) * C_IN];
        v8bf a0lo = *(const v8bf*)(wp0 + ca0);
        v8bf a0hi = *(const v8bf*)(wp0 + ca1);
        v8bf a1lo = *(const v8bf*)(wp1 + ca0);
        v8bf a1hi = *(const v8bf*)(wp1 + ca1);
        v16bf a0 = __builtin_shufflevector(a0lo, a0hi,
                     0,1,2,3,4,5,6,7,8,9,10,11,12,13,14,15);
        v16bf a1 = __builtin_shufflevector(a1lo, a1hi,
                     0,1,2,3,4,5,6,7,8,9,10,11,12,13,14,15);

        v16bf b0 = *(const v16bf*)&sX[(r * LDS_W + wb0 + l + s) * C_IN + cb];
        v16bf b1 = *(const v16bf*)&sX[(r * LDS_W + wb1 + l + s) * C_IN + cb];

        acc00 = __builtin_amdgcn_wmma_f32_16x16x32_bf16(
                    false, a0, false, b0, (short)0, acc00, false, false);
        acc10 = __builtin_amdgcn_wmma_f32_16x16x32_bf16(
                    false, a1, false, b0, (short)0, acc10, false, false);
        acc01 = __builtin_amdgcn_wmma_f32_16x16x32_bf16(
                    false, a0, false, b1, (short)0, acc01, false, false);
        acc11 = __builtin_amdgcn_wmma_f32_16x16x32_bf16(
                    false, a1, false, b1, (short)0, acc11, false, false);
    }

    // --- epilogue: bias + store. f32 C/D layout: VGPR i -> M = i + 8*half --
    #pragma unroll
    for (int i = 0; i < 8; ++i) {
        const int oc0 = i + half * 8;
        const int oc1 = 16 + i + half * 8;
        const float bv0 = bias[oc0];
        const float bv1 = bias[oc1];
        float* y0 = &Y[((n * C_OUT + oc0) * HW + h) * HW];
        float* y1 = &Y[((n * C_OUT + oc1) * HW + h) * HW];
        y0[wb0 + l] = acc00[i] + bv0;
        y1[wb0 + l] = acc10[i] + bv1;
        y0[wb1 + l] = acc01[i] + bv0;
        y1[wb1 + l] = acc11[i] + bv1;
    }
}

extern "C" void kernel_launch(void* const* d_in, const int* in_sizes, int n_in,
                              void* d_out, int out_size, void* d_ws, size_t ws_size,
                              hipStream_t stream) {
    const float* X = (const float*)d_in[0];   // (32,32,224,224) fp32
    const float* W = (const float*)d_in[1];   // (32,32,3,3) fp32
    const float* b = (const float*)d_in[2];   // (32,) fp32
    float*       Y = (float*)d_out;           // (32,32,224,224) fp32
    __bf16*     Wp = (__bf16*)d_ws;           // 18432 B scratch

    repack_weights<<<dim3((W_ELEMS + 255) / 256), 256, 0, stream>>>(W, Wp);
    conv_wmma_bf16<<<dim3(32 * HW), 224, 0, stream>>>(X, Wp, b, Y);
}
